// FactorizedRandomAttention_60447369724634
// MI455X (gfx1250) — compile-verified, matrix operand order checked by reading, not run on hardware
//
#include <hip/hip_runtime.h>

typedef __attribute__((ext_vector_type(16))) _Float16 v16h;
typedef __attribute__((ext_vector_type(8)))  _Float16 h8;
typedef __attribute__((ext_vector_type(8)))  float    v8f;

constexpr int kB = 8, kH = 8, kS = 2048, kF = 64, kDV = 64;
constexpr float kNEG  = -1000000000.0f;
constexpr float LOG2E = 1.44269504088896340736f;

// ---- packed workspace geometry ----
constexpr int NA2  = kB * kH * (kS / 16) * 2 * 32;       // a2 frag chunks: 524288 x 16 halves
constexpr int NV   = kB * kH * (kS / 32) * 4 * 32;       // v  frag chunks: 524288 x 16 halves
constexpr int NM32 = kB * kS * (kS / 32);                // mask words (u32/32cols): 1048576
constexpr size_t A2H_BYTES = (size_t)NA2 * 16 * 2;       // 16 MiB
constexpr size_t VH_BYTES  = (size_t)NV  * 16 * 2;       // 16 MiB
constexpr size_t MPK_BYTES = (size_t)NM32 * 4;           // 4 MiB
constexpr size_t WS_NEEDED = A2H_BYTES + VH_BYTES + MPK_BYTES;

// ================= prepass: pack a2/v to f16 fragment order, bit-pack mask (u32 per 32 cols) =================
__global__ __launch_bounds__(256) void pack_kernel(
    const float* __restrict__ v, const int* __restrict__ mask,
    const float* __restrict__ a2,
    _Float16* __restrict__ a2h, _Float16* __restrict__ vh,
    unsigned* __restrict__ mpk)
{
    const int tid = blockIdx.x * 256 + threadIdx.x;
    if (tid < NA2) {
        const int lane = tid & 31, f = (tid >> 5) & 1, t = (tid >> 6) & 127, bh = tid >> 13;
        const int g = lane >> 4, n = lane & 15, ko = f * 32, kb = 8 * g;
        const float* src = a2 + (size_t)bh * kF * kS + t * 16 + n;
        _Float16* dst = a2h + (size_t)tid * 16;
        #pragma unroll
        for (int j = 0; j < 8; ++j) {
            dst[j]     = (_Float16)src[(size_t)(ko + kb + j)      * kS];
            dst[j + 8] = (_Float16)src[(size_t)(ko + 16 + kb + j) * kS];
        }
    } else if (tid < NA2 + NV) {
        const int id = tid - NA2;
        const int lane = id & 31, nt = (id >> 5) & 3, tp = (id >> 7) & 63, bh = id >> 13;
        const int g = lane >> 4, n = lane & 15, kb = 8 * g;
        const float* src = v + (size_t)bh * kS * kDV + (size_t)tp * 32 * kDV + nt * 16 + n;
        _Float16* dst = vh + (size_t)id * 16;
        #pragma unroll
        for (int j = 0; j < 8; ++j) {
            dst[j]     = (_Float16)src[(size_t)(kb + j)      * kDV];
            dst[j + 8] = (_Float16)src[(size_t)(16 + kb + j) * kDV];
        }
    } else if (tid < NA2 + NV + NM32) {
        // mpk[(b*kS + row)*64 + w] : bit c = (mask[b,row,w*32+c] != 0)
        const int id = tid - NA2 - NV;
        const int* m = mask + (size_t)id * 32;
        unsigned w = 0;
        #pragma unroll
        for (int q = 0; q < 32; ++q)
            w |= (__builtin_nontemporal_load(m + q) != 0 ? 1u : 0u) << q;
        mpk[id] = w;
    }
}

// ================= main kernel: 3-pass factorized attention, 16 rows per wave =================
template <bool PACKED>
__global__ __launch_bounds__(128) void fra_kernel(
    const float* __restrict__ v, const int* __restrict__ mask,
    const float* __restrict__ a1, const float* __restrict__ a2,
    const _Float16* __restrict__ a2h, const _Float16* __restrict__ vh,
    const unsigned* __restrict__ mpk,
    float* __restrict__ out, float* __restrict__ attn)
{
    __shared__ alignas(16) _Float16 lds_p[4][16 * 32];   // per-wave p staging
    __shared__ alignas(16) unsigned lds_m[4][16 * 64];   // per-wave mask-word cache (4 KB)

    const int wave = threadIdx.x >> 5;
    const int lane = threadIdx.x & 31;
    const int g    = lane >> 4;
    const int ln   = lane & 15;
    const int kb   = g * 8;

    const int wg   = blockIdx.x;
    const int bh   = wg / (kS / 64);
    const int row0 = (wg % (kS / 64)) * 64 + wave * 16;
    const int b    = bh / kH;

    const float* a1p  = a1 + (size_t)bh * kS * kF + (size_t)row0 * kF;
    const float* a2p  = a2 + (size_t)bh * kF * kS;
    const float* vp   = v  + (size_t)bh * kS * kDV;
    const int*   mp   = mask + (size_t)b * kS * kS + (size_t)row0 * kS;
    float* outp  = out  + (size_t)bh * kS * kDV + (size_t)row0 * kDV;
    float* attnp = attn + (size_t)bh * (size_t)kS * kS + (size_t)row0 * kS;
    _Float16* plds = &lds_p[wave][0];

    const v16h* a2fv = (const v16h*)a2h + ((size_t)bh * 128) * 64 + lane;  // +t*64 +f*32
    const v16h* vfv  = (const v16h*)vh  + ((size_t)bh * 64) * 128 + lane;  // +tp*128 +nt*32

    // ---- mask cache: copy this wave's 4 KB of packed mask words into LDS ----
    const unsigned* mw = nullptr;
    if constexpr (PACKED) {
        const uint4* src = (const uint4*)(mpk + ((size_t)b * kS + row0) * 64);
        uint4* dst = (uint4*)&lds_m[wave][0];
        #pragma unroll
        for (int i = 0; i < 8; ++i) dst[i * 32 + lane] = src[i * 32 + lane];
        asm volatile("s_wait_dscnt 0" ::: "memory");
        mw = &lds_m[wave][0];
    }

    // ---- persistent A fragments: a1 rows (16x64 f32 -> 2x f16 16x32) ----
    v16h afrag[2];
    {
        const float* arow = a1p + (size_t)ln * kF;
        #pragma unroll
        for (int f = 0; f < 2; ++f)
            #pragma unroll
            for (int j = 0; j < 8; ++j) {
                afrag[f][j]     = (_Float16)arow[f * 32 + kb + j];
                afrag[f][j + 8] = (_Float16)arow[f * 32 + 16 + kb + j];
            }
    }

    // raw score tile at column t0 (no mask): 2 WMMAs over K=64
    auto score_raw = [&](int t0) -> v8f {
        v16h b0, b1;
        if constexpr (PACKED) {
            const v16h* base = a2fv + (size_t)(t0 >> 4) * 64;
            b0 = base[0];
            b1 = base[32];
        } else {
            #pragma unroll
            for (int j = 0; j < 8; ++j) {
                b0[j]     = (_Float16)a2p[(size_t)(kb + j)      * kS + t0 + ln];
                b0[j + 8] = (_Float16)a2p[(size_t)(16 + kb + j) * kS + t0 + ln];
                b1[j]     = (_Float16)a2p[(size_t)(32 + kb + j) * kS + t0 + ln];
                b1[j + 8] = (_Float16)a2p[(size_t)(48 + kb + j) * kS + t0 + ln];
            }
        }
        v8f c = {};
        c = __builtin_amdgcn_wmma_f32_16x16x32_f16(false, afrag[0], false, b0,
                                                   (short)0, c, false, false);
        c = __builtin_amdgcn_wmma_f32_16x16x32_f16(false, afrag[1], false, b1,
                                                   (short)0, c, false, false);
        return c;
    };

    // mask a 32-col pair of tiles; one word per row (from LDS cache) covers both
    auto mask_pair = [&](int t0, v8f& s0, v8f& s1) {
        #pragma unroll
        for (int r = 0; r < 8; ++r) {
            bool k0, k1;
            if constexpr (PACKED) {
                const unsigned w = mw[(r + 8 * g) * 64 + (t0 >> 5)];   // LDS broadcast
                k0 = ((w >> ln) & 1u) != 0u;
                k1 = ((w >> (16 + ln)) & 1u) != 0u;
            } else {
                k0 = mp[(size_t)(r + 8 * g) * kS + t0 + ln] != 0;
                k1 = mp[(size_t)(r + 8 * g) * kS + t0 + 16 + ln] != 0;
            }
            s0[r] = k0 ? s0[r] : kNEG;
            s1[r] = k1 ? s1[r] : kNEG;
        }
    };

    // ---- Pass 0: exact row max, shuffle-free inner loop ----
    float m_r[8];
    #pragma unroll
    for (int r = 0; r < 8; ++r) m_r[r] = -__builtin_inff();

    for (int t0 = 0; t0 < kS; t0 += 32) {
        v8f s0 = score_raw(t0);
        v8f s1 = score_raw(t0 + 16);
        mask_pair(t0, s0, s1);
        #pragma unroll
        for (int r = 0; r < 8; ++r)
            m_r[r] = fmaxf(m_r[r], fmaxf(s0[r], s1[r]));
    }
    // one butterfly per row for the whole wave
    #pragma unroll
    for (int r = 0; r < 8; ++r) {
        float mx = m_r[r];
        mx = fmaxf(mx, __shfl_xor(mx, 1));
        mx = fmaxf(mx, __shfl_xor(mx, 2));
        mx = fmaxf(mx, __shfl_xor(mx, 4));
        mx = fmaxf(mx, __shfl_xor(mx, 8));
        m_r[r] = mx;                          // group-uniform, final
    }

    // ---- Pass 1: e = exp(s - m), per-lane l, out = p @ v (no rescaling) ----
    float lpart[8];
    v8f oacc[4];
    #pragma unroll
    for (int r = 0; r < 8; ++r) lpart[r] = 0.0f;
    #pragma unroll
    for (int nt = 0; nt < 4; ++nt) oacc[nt] = v8f{};

    for (int t0 = 0; t0 < kS; t0 += 32) {
        v8f s0 = score_raw(t0);
        v8f s1 = score_raw(t0 + 16);
        mask_pair(t0, s0, s1);

        #pragma unroll
        for (int r = 0; r < 8; ++r) {
            const float e0 = exp2f((s0[r] - m_r[r]) * LOG2E);
            const float e1 = exp2f((s1[r] - m_r[r]) * LOG2E);
            lpart[r] += e0 + e1;
            plds[(r + 8 * g) * 32 + ln]      = (_Float16)e0;   // e <= 1, safe in f16
            plds[(r + 8 * g) * 32 + 16 + ln] = (_Float16)e1;
        }
        asm volatile("s_wait_dscnt 0" ::: "memory");

        v16h pa;
        {
            const h8* prow = (const h8*)(plds + ln * 32);
            h8 c0 = prow[g];
            h8 c1 = prow[2 + g];
            #pragma unroll
            for (int j = 0; j < 8; ++j) { pa[j] = c0[j]; pa[j + 8] = c1[j]; }
        }
        asm volatile("" ::: "memory");

        #pragma unroll
        for (int nt = 0; nt < 4; ++nt) {
            v16h bv;
            if constexpr (PACKED) {
                bv = vfv[(size_t)(t0 >> 5) * 128 + nt * 32];
            } else {
                const float* vcol = vp + (size_t)t0 * kDV + nt * 16 + ln;
                #pragma unroll
                for (int j = 0; j < 8; ++j) {
                    bv[j]     = (_Float16)vcol[(size_t)(kb + j)      * kDV];
                    bv[j + 8] = (_Float16)vcol[(size_t)(16 + kb + j) * kDV];
                }
            }
            oacc[nt] = __builtin_amdgcn_wmma_f32_16x16x32_f16(
                false, pa, false, bv, (short)0, oacc[nt], false, false);
        }
    }

    // ---- final l: one 16-lane reduction per row ----
    float il_r[8];
    #pragma unroll
    for (int r = 0; r < 8; ++r) {
        float s = lpart[r];
        s += __shfl_xor(s, 1);
        s += __shfl_xor(s, 2);
        s += __shfl_xor(s, 4);
        s += __shfl_xor(s, 8);
        il_r[r] = 1.0f / s;
    }

    // ---- write out = oacc / l (non-temporal: streaming output) ----
    #pragma unroll
    for (int r = 0; r < 8; ++r)
        #pragma unroll
        for (int nt = 0; nt < 4; ++nt)
            __builtin_nontemporal_store(oacc[nt][r] * il_r[r],
                outp + (size_t)(r + 8 * g) * kDV + nt * 16 + ln);

    // ---- Pass 2: recompute scores (L2-hot), write attn once, non-temporal ----
    for (int t0 = 0; t0 < kS; t0 += 32) {
        v8f s0 = score_raw(t0);
        v8f s1 = score_raw(t0 + 16);
        mask_pair(t0, s0, s1);
        #pragma unroll
        for (int r = 0; r < 8; ++r) {
            const float p0 = exp2f((s0[r] - m_r[r]) * LOG2E) * il_r[r];
            const float p1 = exp2f((s1[r] - m_r[r]) * LOG2E) * il_r[r];
            float* arow = attnp + (size_t)(r + 8 * g) * kS + t0 + ln;
            __builtin_nontemporal_store(p0, arow);
            __builtin_nontemporal_store(p1, arow + 16);
        }
    }
}

extern "C" void kernel_launch(void* const* d_in, const int* in_sizes, int n_in,
                              void* d_out, int out_size, void* d_ws, size_t ws_size,
                              hipStream_t stream) {
    const float* v    = (const float*)d_in[0];
    const int*   mask = (const int*)d_in[1];
    const float* a1   = (const float*)d_in[2];
    const float* a2   = (const float*)d_in[3];
    float* out  = (float*)d_out;
    float* attn = out + (size_t)kB * kH * kS * kDV;   // tuple: (out, attn) flat

    dim3 grid(kB * kH * (kS / 64));
    dim3 block(128);

    if (ws_size >= WS_NEEDED) {
        char* ws = (char*)d_ws;
        _Float16* a2h = (_Float16*)ws;
        _Float16* vh  = (_Float16*)(ws + A2H_BYTES);
        unsigned* mpk = (unsigned*)(ws + A2H_BYTES + VH_BYTES);
        const int ntot = NA2 + NV + NM32;
        hipLaunchKernelGGL(pack_kernel, dim3((ntot + 255) / 256), dim3(256), 0, stream,
                           v, mask, a2, a2h, vh, mpk);
        hipLaunchKernelGGL((fra_kernel<true>), grid, block, 0, stream,
                           v, mask, a1, a2, a2h, vh, mpk, out, attn);
    } else {
        hipLaunchKernelGGL((fra_kernel<false>), grid, block, 0, stream,
                           v, mask, a1, a2, (const _Float16*)nullptr,
                           (const _Float16*)nullptr, (const unsigned*)nullptr,
                           out, attn);
    }
}
